// GAT_26560077758615
// MI455X (gfx1250) — matrix operand order repaired and checked by reading, not compiled, not run on hardware
//
#include <hip/hip_runtime.h>

// ---------------------------------------------------------------------------
// GAT, 4 layers, N=8192, dims 512->256->64->256->512, fused rank-1 attention.
// All heavy GEMMs via v_wmma_f32_16x16x32_bf16 (wave32, gfx1250).
// gemm_h emits H^T (bf16) so attention B-fragments are direct global loads.
// ---------------------------------------------------------------------------

typedef __bf16 bf16;
typedef __attribute__((ext_vector_type(16))) __bf16 v16bf;
typedef __attribute__((ext_vector_type(8)))  float  v8f;

struct alignas(16) U128 { unsigned int x, y, z, w; };
struct alignas(8)  U64p { unsigned int x, y; };
struct U256 { U128 a, b; };

static __device__ __forceinline__ unsigned short f2b(float f) {
  return __builtin_bit_cast(unsigned short, (bf16)f);
}

static __device__ __forceinline__ v8f wmma_bf16(v16bf a, v16bf b, v8f c) {
  // D = A(16x32 bf16) x B(32x16 bf16) + C(16x16 f32)
  return __builtin_amdgcn_wmma_f32_16x16x32_bf16(false, a, false, b,
                                                 (short)0, c, false, false);
}

// ---------------------------------------------------------------------------
// Pack adj (int32 {0,1}) into bitmask: word w bit j = adj[w*32+j] != 0.
// 268 MB read once; 8 MB bitmask stays hot in L2 for all 4 layers.
// ---------------------------------------------------------------------------
__global__ __launch_bounds__(256) void pack_adj_kernel(
    const int* __restrict__ adj, unsigned* __restrict__ bits)
{
  size_t w = (size_t)blockIdx.x * 256 + threadIdx.x;
  const int* p = adj + w * 32;
  unsigned m = 0;
#pragma unroll
  for (int q = 0; q < 8; ++q) {
    U128 v = *(const U128*)(p + q * 4);
    if (v.x) m |= 1u << (q * 4 + 0);
    if (v.y) m |= 1u << (q * 4 + 1);
    if (v.z) m |= 1u << (q * 4 + 2);
    if (v.w) m |= 1u << (q * 4 + 3);
  }
  bits[w] = m;
}

// ---------------------------------------------------------------------------
// h = X @ W + b  (X fp32 [nrows,din], W fp32 [din,dout]).
// Outputs: Hf fp32 [nrows,dout]  and  HbT bf16 [dout,nrows] (transposed).
// WG = 256 thr (8 waves), tile 128(M) x 64(N), K-step 32, bf16 WMMA.
// ---------------------------------------------------------------------------
__global__ __launch_bounds__(256) void gemm_h_kernel(
    const float* __restrict__ X, const float* __restrict__ W,
    const float* __restrict__ bias, float* __restrict__ Hf,
    bf16* __restrict__ HbT, int din, int dout, int nrows)
{
  __shared__ bf16 xA[128 * 40];   // row-major [m][k], stride 40 (bank-friendly)
  __shared__ bf16 Wt[64 * 40];    // transposed [n][k]
  __shared__ bf16 Ht[128 * 72];   // C tile bounce buffer [m][n] for transpose
  const int tid  = threadIdx.x;
  const int lane = tid & 31, wv = tid >> 5;
  const int half = lane >> 4, lm = lane & 15;
  const int m0 = blockIdx.y * 128;
  const int n0 = blockIdx.x * 64;

  v8f acc[4];
#pragma unroll
  for (int t = 0; t < 4; ++t) acc[t] = (v8f){0,0,0,0,0,0,0,0};

  for (int k0 = 0; k0 < din; k0 += 32) {
    __syncthreads();
    // X tile 128x32 f32 -> bf16 LDS (4 float4 per thread)
#pragma unroll
    for (int q = 0; q < 4; ++q) {
      int f = tid * 4 + q;
      int r = f >> 3, kq = f & 7;
      float4 v = *(const float4*)(X + (size_t)(m0 + r) * din + k0 + kq * 4);
      U64p pk;
      pk.x = (unsigned)f2b(v.x) | ((unsigned)f2b(v.y) << 16);
      pk.y = (unsigned)f2b(v.z) | ((unsigned)f2b(v.w) << 16);
      *(U64p*)&xA[r * 40 + kq * 4] = pk;
    }
    // W tile 32x64 f32 -> transposed bf16 LDS (2 float4 per thread)
    unsigned short* wt = (unsigned short*)Wt;
#pragma unroll
    for (int q = 0; q < 2; ++q) {
      int f = tid * 2 + q;
      int kk = f >> 4, n4 = f & 15;
      float4 v = *(const float4*)(W + (size_t)(k0 + kk) * dout + n0 + n4 * 4);
      wt[(n4 * 4 + 0) * 40 + kk] = f2b(v.x);
      wt[(n4 * 4 + 1) * 40 + kk] = f2b(v.y);
      wt[(n4 * 4 + 2) * 40 + kk] = f2b(v.z);
      wt[(n4 * 4 + 3) * 40 + kk] = f2b(v.w);
    }
    __syncthreads();
    // A fragment: wave wv owns rows 16*wv..; lane half selects K base {0,8}
    U256 ta;
    int am = wv * 16 + lm;
    ta.a = *(const U128*)&xA[am * 40 + half * 8];
    ta.b = *(const U128*)&xA[am * 40 + half * 8 + 16];
    v16bf af = __builtin_bit_cast(v16bf, ta);
#pragma unroll
    for (int ct = 0; ct < 4; ++ct) {
      int bc = ct * 16 + lm;
      U256 tb;
      tb.a = *(const U128*)&Wt[bc * 40 + half * 8];
      tb.b = *(const U128*)&Wt[bc * 40 + half * 8 + 16];
      v16bf bfm = __builtin_bit_cast(v16bf, tb);
      acc[ct] = wmma_bf16(af, bfm, acc[ct]);
    }
  }
  // epilogue: bias, fp32 store, and bf16 tile -> LDS for transposed store
  __syncthreads();
  {
    unsigned short* ht = (unsigned short*)Ht;
#pragma unroll
    for (int ct = 0; ct < 4; ++ct) {
      int gn = n0 + ct * 16 + lm;
      float bv = bias[gn];
#pragma unroll
      for (int e = 0; e < 8; ++e) {
        int mrow = wv * 16 + e + half * 8;
        float val = acc[ct][e] + bv;
        Hf[(size_t)(m0 + mrow) * dout + gn] = val;
        ht[mrow * 72 + ct * 16 + lm] = f2b(val);
      }
    }
  }
  __syncthreads();
  // transposed write: thread -> (col n, 32-row segment), coalesced 16B stores
  {
    const unsigned short* ht = (const unsigned short*)Ht;
    int n = tid >> 2, seg = tid & 3;
#pragma unroll
    for (int g = 0; g < 4; ++g) {
      int mb = seg * 32 + g * 8;
      unsigned e0 = ht[(mb + 0) * 72 + n], e1 = ht[(mb + 1) * 72 + n];
      unsigned e2 = ht[(mb + 2) * 72 + n], e3 = ht[(mb + 3) * 72 + n];
      unsigned e4 = ht[(mb + 4) * 72 + n], e5 = ht[(mb + 5) * 72 + n];
      unsigned e6 = ht[(mb + 6) * 72 + n], e7 = ht[(mb + 7) * 72 + n];
      U128 pk;
      pk.x = e0 | (e1 << 16);
      pk.y = e2 | (e3 << 16);
      pk.z = e4 | (e5 << 16);
      pk.w = e6 | (e7 << 16);
      *(U128*)(HbT + (size_t)(n0 + n) * nrows + m0 + mb) = pk;
    }
  }
}

// ---------------------------------------------------------------------------
// s[i] = h[i,:]@a_s ; n[i] = h[i,:]@a_n   (one wave per row)
// ---------------------------------------------------------------------------
__global__ __launch_bounds__(256) void sn_kernel(
    const float* __restrict__ H, const float* __restrict__ as_,
    const float* __restrict__ an_, float* __restrict__ S,
    float* __restrict__ Nv, int dout)
{
  int lane = threadIdx.x & 31, wv = threadIdx.x >> 5;
  int row = blockIdx.x * 8 + wv;
  const float* hr = H + (size_t)row * dout;
  float s = 0.f, n = 0.f;
  for (int c = lane; c < dout; c += 32) {
    float h = hr[c];
    s += h * as_[c];
    n += h * an_[c];
  }
#pragma unroll
  for (int off = 16; off; off >>= 1) {
    s += __shfl_xor(s, off, 32);
    n += __shfl_xor(n, off, 32);
  }
  if (lane == 0) { S[row] = s; Nv[row] = n; }
}

// ---------------------------------------------------------------------------
// Fused rank-1 attention + softmax + P@V + ELU.
// scores bounded: |lrelu(cos)| <= 1, so fixed shift M = max(b,-0.2b) gives the
// exact softmax (masked lanes underflow to 0, same as reference).
// WG = 256 thr / 8 waves. Waves partition OUTPUT COLUMNS (the B-dependent
// axis); each wave sweeps NRT row-tiles (all 4 if DOUT>=128) reusing B fragments in registers ->
// no redundant H^T loads within a WG if DOUT>=128. BM=64 rows/WG, BN=32 cols per j-tile.
// B fragments load straight from H^T (bf16 [DOUT,N], L2-resident).
// ---------------------------------------------------------------------------
template <int DOUT>
__global__ __launch_bounds__(256) void attn_kernel(
    const bf16* __restrict__ HbT, const float* __restrict__ S,
    const float* __restrict__ Nv, const unsigned* __restrict__ adjb,
    const float* __restrict__ betap, float* __restrict__ Out, int nodes)
{
  constexpr int CW  = (DOUT >= 128) ? 8 : DOUT / 16;  // col wave-groups
  constexpr int RS  = 8 / CW;                         // row splits
  constexpr int NCT = DOUT / (16 * CW);               // col tiles / wave
  constexpr int NRT = 4 / RS;                         // row tiles / wave
  __shared__ bf16  P[2][64 * 40];                     // double-buffered weights
  __shared__ float rowsum[64];

  const int tid  = threadIdx.x;
  const int lane = tid & 31, wv = tid >> 5;
  const int half = lane >> 4, lm = lane & 15;
  const int cg = wv % CW;            // column group of this wave
  const int rs = wv / CW;            // row split of this wave
  const int i0 = blockIdx.x * 64;

  const float beta = betap[0];
  const float mfix = fmaxf(beta, -0.2f * beta);
  const int   pi  = tid & 63;        // row handled for P compute
  const int   oct = tid >> 6;        // 8-col octet handled for P compute
  const float si  = S[i0 + pi];
  const float asi = fabsf(si);
  const int wstride = nodes >> 5;

  if (tid < 64) rowsum[tid] = 0.f;
  __syncthreads();

  v8f acc[NRT][NCT];
#pragma unroll
  for (int r = 0; r < NRT; ++r)
#pragma unroll
    for (int t = 0; t < NCT; ++t) acc[r][t] = (v8f){0,0,0,0,0,0,0,0};

  const int ntiles = nodes / 32;
  for (int jt = 0; jt < ntiles; ++jt) {
    const int j0 = jt * 32;
    const int par = jt & 1;
    if (jt + 1 < ntiles)
      __builtin_prefetch(HbT + (size_t)(cg * (DOUT / CW)) * nodes + j0 + 32, 0, 1);

    // ---- 8 softmax weights per thread (registers only) ----
    unsigned word = adjb[(size_t)(i0 + pi) * wstride + jt];
    float pv[8], psum = 0.f;
#pragma unroll
    for (int e = 0; e < 8; ++e) {
      int j = oct * 8 + e;
      float nv = Nv[j0 + j];
      float cosv = si * nv / (asi * fabsf(nv) + 1e-7f);
      float mk = cosv > 0.f ? cosv : 0.2f * cosv;        // leaky_relu
      float p = ((word >> j) & 1u) ? __expf(beta * mk - mfix) : 0.f;
      pv[e] = p;
      psum += p;
    }
    U128 pk;
    pk.x = (unsigned)f2b(pv[0]) | ((unsigned)f2b(pv[1]) << 16);
    pk.y = (unsigned)f2b(pv[2]) | ((unsigned)f2b(pv[3]) << 16);
    pk.z = (unsigned)f2b(pv[4]) | ((unsigned)f2b(pv[5]) << 16);
    pk.w = (unsigned)f2b(pv[6]) | ((unsigned)f2b(pv[7]) << 16);
    *(U128*)&P[par][pi * 40 + oct * 8] = pk;
    atomicAdd(&rowsum[pi], psum);
    __syncthreads();

    // ---- B fragments: direct from H^T global (L2), one set per wave ----
    v16bf bfr[NCT];
#pragma unroll
    for (int t = 0; t < NCT; ++t) {
      int c = cg * (DOUT / CW) + t * 16 + lm;
      U256 tb;
      tb.a = *(const U128*)(HbT + (size_t)c * nodes + j0 + half * 8);
      tb.b = *(const U128*)(HbT + (size_t)c * nodes + j0 + half * 8 + 16);
      bfr[t] = __builtin_bit_cast(v16bf, tb);
    }
    // ---- sweep row tiles, reuse each B fragment across NRT WMMAs ----
#pragma unroll
    for (int rt = 0; rt < NRT; ++rt) {
      int am = (rs * NRT + rt) * 16 + lm;
      U256 ta;
      ta.a = *(const U128*)&P[par][am * 40 + half * 8];
      ta.b = *(const U128*)&P[par][am * 40 + half * 8 + 16];
      v16bf af = __builtin_bit_cast(v16bf, ta);
#pragma unroll
      for (int t = 0; t < NCT; ++t)
        acc[rt][t] = wmma_bf16(af, bfr[t], acc[rt][t]);
    }
  }
  __syncthreads();
  // ---- normalize by row sum, ELU, store ----
#pragma unroll
  for (int rt = 0; rt < NRT; ++rt) {
#pragma unroll
    for (int e = 0; e < 8; ++e) {
      int m = (rs * NRT + rt) * 16 + e + half * 8;
      float sum = rowsum[m];
      float inv = sum > 0.f ? 1.f / sum : 0.f;
#pragma unroll
      for (int t = 0; t < NCT; ++t) {
        int c = cg * (DOUT / CW) + t * 16 + lm;
        float v = acc[rt][t][e] * inv;
        v = v > 0.f ? v : (__expf(v) - 1.f);             // ELU
        Out[(size_t)(i0 + m) * DOUT + c] = v;
      }
    }
  }
}

// ---------------------------------------------------------------------------
// z = h / max(||h||_2, 1e-12), rows of [N,64]; fp32 out + bf16 copy for zz^T.
// ---------------------------------------------------------------------------
__global__ __launch_bounds__(256) void norm_kernel(
    const float* __restrict__ H, float* __restrict__ Zf, bf16* __restrict__ Zb)
{
  int lane = threadIdx.x & 31, wv = threadIdx.x >> 5;
  int row = blockIdx.x * 8 + wv;
  const float* hr = H + (size_t)row * 64;
  float a = hr[lane], b = hr[lane + 32];
  float ss = a * a + b * b;
#pragma unroll
  for (int off = 16; off; off >>= 1) ss += __shfl_xor(ss, off, 32);
  float sc = 1.f / fmaxf(sqrtf(ss), 1e-12f);
  float za = a * sc, zb2 = b * sc;
  Zf[(size_t)row * 64 + lane]      = za;
  Zf[(size_t)row * 64 + lane + 32] = zb2;
  Zb[(size_t)row * 64 + lane]      = (bf16)za;
  Zb[(size_t)row * 64 + lane + 32] = (bf16)zb2;
}

// ---------------------------------------------------------------------------
// A_pred = sigmoid(Z @ Z^T), Z bf16 [N,64] (1 MB, L2-hot). Both fragments map
// directly onto 16B global chunks (B's K axis is Z's contiguous feature axis).
// WG tile 128x128, 8 waves each 16 rows x 128 cols.
// ---------------------------------------------------------------------------
__global__ __launch_bounds__(256) void apred_kernel(
    const bf16* __restrict__ Z, float* __restrict__ Out, int n)
{
  const int tid  = threadIdx.x;
  const int lane = tid & 31, wv = tid >> 5;
  const int half = lane >> 4, lm = lane & 15;
  const int i0 = blockIdx.y * 128, j0 = blockIdx.x * 128;

  v8f acc[8];
#pragma unroll
  for (int t = 0; t < 8; ++t) acc[t] = (v8f){0,0,0,0,0,0,0,0};

#pragma unroll
  for (int k0 = 0; k0 < 64; k0 += 32) {
    int row = i0 + wv * 16 + lm;
    U256 ta;
    ta.a = *(const U128*)(Z + (size_t)row * 64 + k0 + half * 8);
    ta.b = *(const U128*)(Z + (size_t)row * 64 + k0 + half * 8 + 16);
    v16bf af = __builtin_bit_cast(v16bf, ta);
#pragma unroll
    for (int t = 0; t < 8; ++t) {
      int col = j0 + t * 16 + lm;
      U256 tb;
      tb.a = *(const U128*)(Z + (size_t)col * 64 + k0 + half * 8);
      tb.b = *(const U128*)(Z + (size_t)col * 64 + k0 + half * 8 + 16);
      v16bf bfm = __builtin_bit_cast(v16bf, tb);
      acc[t] = wmma_bf16(af, bfm, acc[t]);
    }
  }
#pragma unroll
  for (int t = 0; t < 8; ++t) {
    int col = j0 + t * 16 + lm;
#pragma unroll
    for (int e = 0; e < 8; ++e) {
      int row = i0 + wv * 16 + e + half * 8;
      float v = acc[t][e];
      Out[(size_t)row * n + col] = 1.f / (1.f + __expf(-v));
    }
  }
}

// ---------------------------------------------------------------------------
extern "C" void kernel_launch(void* const* d_in, const int* in_sizes, int n_in,
                              void* d_out, int out_size, void* d_ws,
                              size_t ws_size, hipStream_t stream)
{
  (void)in_sizes; (void)n_in; (void)out_size; (void)ws_size;
  const int N = 8192, F = 512, Hd = 256, E = 64;

  const float* x  = (const float*)d_in[0];
  const int*  adj = (const int*)d_in[1];
  const float *W[4], *bb[4], *as_[4], *an_[4], *bt[4];
  for (int i = 0; i < 4; ++i) {
    W[i]   = (const float*)d_in[2 + 5 * i];
    bb[i]  = (const float*)d_in[3 + 5 * i];
    as_[i] = (const float*)d_in[4 + 5 * i];
    an_[i] = (const float*)d_in[5 + 5 * i];
    bt[i]  = (const float*)d_in[6 + 5 * i];
  }

  char* ws = (char*)d_ws;
  unsigned* adjb = (unsigned*)ws;                                   // 8 MB
  float* bufA = (float*)(ws + (size_t)8  * 1024 * 1024);            // h  16 MB
  float* bufB = (float*)(ws + (size_t)24 * 1024 * 1024);            // hp 16 MB
  bf16*  hbT  = (bf16*) (ws + (size_t)40 * 1024 * 1024);            // H^T 8 MB
  float* sS   = (float*)(ws + (size_t)48 * 1024 * 1024);            // 32 KB
  float* nS   = (float*)(ws + (size_t)48 * 1024 * 1024 + 64 * 1024);
  bf16*  zb   = (bf16*) (ws + (size_t)48 * 1024 * 1024 + 128 * 1024); // 1 MB

  float* outA = (float*)d_out;                 // A_pred [N,N]
  float* outZ = outA + (size_t)N * N;          // z      [N,E]
  float* outX = outZ + (size_t)N * E;          // x_     [N,F]

  pack_adj_kernel<<<(N / 32) * (N / 256), 256, 0, stream>>>(adj, adjb);

  // ---- layer 1: 512 -> 256 ----
  gemm_h_kernel<<<dim3(Hd / 64, N / 128), 256, 0, stream>>>(x, W[0], bb[0], bufA, hbT, F, Hd, N);
  sn_kernel<<<N / 8, 256, 0, stream>>>(bufA, as_[0], an_[0], sS, nS, Hd);
  attn_kernel<256><<<N / 64, 256, 0, stream>>>(hbT, sS, nS, adjb, bt[0], bufB, N);

  // ---- layer 2: 256 -> 64 ----
  gemm_h_kernel<<<dim3(E / 64, N / 128), 256, 0, stream>>>(bufB, W[1], bb[1], bufA, hbT, Hd, E, N);
  sn_kernel<<<N / 8, 256, 0, stream>>>(bufA, as_[1], an_[1], sS, nS, E);
  attn_kernel<64><<<N / 64, 256, 0, stream>>>(hbT, sS, nS, adjb, bt[1], bufB, N);

  // ---- z = normalize(h2) ----
  norm_kernel<<<N / 8, 256, 0, stream>>>(bufB, outZ, zb);

  // ---- layer 3: 64 -> 256 (input z) ----
  gemm_h_kernel<<<dim3(Hd / 64, N / 128), 256, 0, stream>>>(outZ, W[2], bb[2], bufA, hbT, E, Hd, N);
  sn_kernel<<<N / 8, 256, 0, stream>>>(bufA, as_[2], an_[2], sS, nS, Hd);
  attn_kernel<256><<<N / 64, 256, 0, stream>>>(hbT, sS, nS, adjb, bt[2], bufB, N);

  // ---- layer 4: 256 -> 512 ----
  gemm_h_kernel<<<dim3(F / 64, N / 128), 256, 0, stream>>>(bufB, W[3], bb[3], bufA, hbT, Hd, F, N);
  sn_kernel<<<N / 8, 256, 0, stream>>>(bufA, as_[3], an_[3], sS, nS, F);
  attn_kernel<512><<<N / 64, 256, 0, stream>>>(hbT, sS, nS, adjb, bt[3], outX, N);

  // ---- A_pred = sigmoid(z z^T) ----
  apred_kernel<<<dim3(N / 128, N / 128), 256, 0, stream>>>(zb, outA, N);
}
